// Qwen3VLLaCTSWIGLULayer_44555990729465
// MI455X (gfx1250) — compile-verified, matrix operand order checked by reading, not run on hardware
//
#include <hip/hip_runtime.h>
#include <hip/hip_bf16.h>

typedef __attribute__((ext_vector_type(16))) __bf16 v16bf;
typedef __attribute__((ext_vector_type(8)))  __bf16 v8bf;
typedef __attribute__((ext_vector_type(4)))  __bf16 v4bf;
typedef __attribute__((ext_vector_type(8)))  float  v8f;

#define BH_N   32
#define SEQ    8192
#define DIM    128
#define CHUNK  2048
#define NCHUNK (SEQ / CHUNK)
#define CTILE  32
#define NITER  (CHUNK / CTILE)
#define NTHR   512
#define EPSV   1e-5f

// ---- LDS layout (bytes) ----
// sW0  bf16[128*128] @ 0       natural [h][d]  (B N-major for gate)
// sW2  bf16[128*128] @ 32768   natural [h][d]  (B N-major for hid_pre)
// sW1T bf16[128*128] @ 65536   [h][d]          (B N-major for dhidden)
// sK   bf16[32*128]  @ 98304   [c][d] (A gate; q-phase: q tile)
// sV   bf16[32*128]  @ 106496  [c][d] (A dhidden; q-phase: h natural = B for o^T)
// sKT  bf16[128*32]  @ 114688  [d][c] (B N-major for dw0/dw2)
// sVT  bf16[128*32]  @ 122880  [d][c] (B N-major for dw1^T)
// sDgT bf16[128*32]  @ 131072  [h][c] (A dw0 : dgate^T * lr0)
// sDpT bf16[128*32]  @ 139264  [h][c] (A dw2 : dhidpre^T * lr2)
// sHT  bf16[128*32]  @ 147456  [h][c] (A dw1^T: hidden^T * lr1)
// sOT  f32[128*32]   alias @139264 (sDpT+sHT; q-phase o^T staging)
// misc floats        @ 155648
#define SMEM_BYTES 159648

// Fragment loader: per-lane row (row0+ln), two contiguous 8-elem K groups at
// k0 and k0+16 (plus hi*8).  Used for A and for N-major-stored B operands.
// Lowers to 2x ds_load_b128.
__device__ __forceinline__ v16bf load_frag(const __bf16* __restrict__ p, int S,
                                           int row0, int k0, int ln, int hi) {
  const __bf16* r = p + (row0 + ln) * S + k0 + hi * 8;
  v8bf lo = *(const v8bf*)(r);
  v8bf hh = *(const v8bf*)(r + 16);
  return __builtin_shufflevector(lo, hh, 0, 1, 2, 3, 4, 5, 6, 7,
                                 8, 9, 10, 11, 12, 13, 14, 15);
}

__device__ __forceinline__ v8f wmma_bf16(v16bf a, v16bf b, v8f c) {
  return __builtin_amdgcn_wmma_f32_16x16x32_bf16(false, a, false, b, (short)0, c,
                                                 false, false);
}

// Per-row momentum + L2-renorm update; 4 threads per row (row = tid>>2).
__device__ __forceinline__ void update_w(float* __restrict__ wm,
                                         float* __restrict__ mprev,
                                         const float* __restrict__ stage,
                                         bool stageT, const float* __restrict__ sNorm,
                                         __bf16* __restrict__ sWdst, bool dstT,
                                         float mm, bool first, int row, int cb,
                                         float* __restrict__ sRed, int tid) {
  float dwt[32];
  float ss = 0.f;
#pragma unroll
  for (int j = 0; j < 32; ++j) {
    int col = cb + j;
    float dv = stageT ? stage[col * DIM + row] : stage[row * DIM + col];
    if (!first) dv += mm * mprev[row * DIM + col];
    mprev[row * DIM + col] = dv;                 // m_new = dw (pre-renorm)
    float wv = wm[row * DIM + col] + dv;
    dwt[j] = wv;
    ss += wv * wv;
  }
  sRed[tid] = ss;
  __syncthreads();
  float n = sqrtf(sRed[row * 4 + 0] + sRed[row * 4 + 1] +
                  sRed[row * 4 + 2] + sRed[row * 4 + 3]);
  float sc = sNorm[row] / fmaxf(n, EPSV);
  __syncthreads();
#pragma unroll
  for (int j = 0; j < 32; ++j) {
    int col = cb + j;
    float wv = dwt[j] * sc;
    wm[row * DIM + col] = wv;
    if (dstT) sWdst[col * DIM + row] = (__bf16)wv;
    else      sWdst[row * DIM + col] = (__bf16)wv;
  }
}

__global__ __launch_bounds__(NTHR, 1) void lact_swiglu_kernel(
    const float* __restrict__ gw0, const float* __restrict__ gw1,
    const float* __restrict__ gw2, const float* __restrict__ gq,
    const float* __restrict__ gk, const float* __restrict__ gv,
    const float* __restrict__ glr0, const float* __restrict__ glr1,
    const float* __restrict__ glr2, const float* __restrict__ gmom,
    float* __restrict__ gout, float* __restrict__ ws) {
  __shared__ __align__(16) char smem[SMEM_BYTES];
  __bf16* sW0  = (__bf16*)smem;
  __bf16* sW2  = sW0 + 16384;
  __bf16* sW1T = sW2 + 16384;
  __bf16* sK   = sW1T + 16384;
  __bf16* sV   = sK + 4096;
  __bf16* sKT  = sV + 4096;
  __bf16* sVT  = sKT + 4096;
  __bf16* sDgT = sVT + 4096;
  __bf16* sDpT = sDgT + 4096;
  __bf16* sHT  = sDpT + 4096;
  float*  sOT  = (float*)sDpT;             // 128*32 f32 alias (q-phase)
  float*  sF   = (float*)(sHT + 4096);
  float* sL0 = sF;                          // 32
  float* sL1 = sL0 + 32;
  float* sL2 = sL1 + 32;
  float* sN0 = sL2 + 32;                    // 128 x3
  float* sN1 = sN0 + 128;
  float* sN2 = sN1 + 128;
  float* sRed = sN2 + 128;                  // 512
  float* sMM  = sRed + 512;

  const int bh  = blockIdx.x;
  const int tid = threadIdx.x;
  const int wid = tid >> 5;
  const int lid = tid & 31;
  const int hi  = lid >> 4;
  const int ln  = lid & 15;

  const float* W0g = gw0 + (size_t)bh * DIM * DIM;
  const float* W1g = gw1 + (size_t)bh * DIM * DIM;
  const float* W2g = gw2 + (size_t)bh * DIM * DIM;

  float* hws = ws + (size_t)bh * 9 * DIM * DIM;
  float* w0m = hws;
  float* w1m = hws + 1 * 16384;
  float* w2m = hws + 2 * 16384;
  float* m0p = hws + 3 * 16384;
  float* m1p = hws + 4 * 16384;
  float* m2p = hws + 5 * 16384;
  float* st0 = hws + 6 * 16384;             // dw0   [h][d]
  float* st1 = hws + 7 * 16384;             // dw1^T [h][d]
  float* st2 = hws + 8 * 16384;             // dw2   [h][d]

  // ---- init: masters -> ws, bf16 copies -> LDS, row-norm targets ----
  for (int i = tid; i < 16384; i += NTHR) {
    float a0 = W0g[i], a1 = W1g[i], a2 = W2g[i];
    w0m[i] = a0; w1m[i] = a1; w2m[i] = a2;
    int r = i >> 7, c = i & 127;
    sW0[i]             = (__bf16)a0;        // natural [h][d]
    sW2[i]             = (__bf16)a2;        // natural [h][d]
    sW1T[c * DIM + r]  = (__bf16)a1;        // [h][d] from [d][h]
  }
  if (tid < 384) {
    int m = tid >> 7, row = tid & 127;
    const float* src = (m == 0) ? W0g : (m == 1) ? W1g : W2g;
    float s = 0.f;
    for (int d = 0; d < DIM; ++d) { float x = src[row * DIM + d]; s += x * x; }
    ((m == 0) ? sN0 : (m == 1) ? sN1 : sN2)[row] = sqrtf(s);
  }
  __syncthreads();

  v8f dw0a[4], dw1a[4], dw2a[4];

  for (int ch = 0; ch < NCHUNK; ++ch) {
    // ---- mm = mean(momentum over chunk) ----
    {
      const float* mp = gmom + (size_t)bh * SEQ + ch * CHUNK;
      float s = 0.f;
#pragma unroll
      for (int j = 0; j < 4; ++j) s += mp[tid * 4 + j];
      sRed[tid] = s;
      __syncthreads();
      for (int st = 256; st > 0; st >>= 1) {
        if (tid < st) sRed[tid] += sRed[tid + st];
        __syncthreads();
      }
      if (tid == 0) sMM[0] = sRed[0] * (1.0f / CHUNK);
      __syncthreads();
    }
#pragma unroll
    for (int t = 0; t < 4; ++t)
#pragma unroll
      for (int r = 0; r < 8; ++r) { dw0a[t][r] = 0.f; dw1a[t][r] = 0.f; dw2a[t][r] = 0.f; }

    // ======== K/V pass ========
    for (int it = 0; it < NITER; ++it) {
      const int rowg = ch * CHUNK + it * CTILE;
      const float4* kg = (const float4*)(gk + ((size_t)bh * SEQ + rowg) * DIM);
      const float4* vg = (const float4*)(gv + ((size_t)bh * SEQ + rowg) * DIM);
#pragma unroll
      for (int j = 0; j < 2; ++j) {
        int f4 = tid + j * NTHR;
        float4 kv = kg[f4];
        float4 vv = vg[f4];
        int base = f4 * 4;
        int c = base >> 7, d = base & 127;
        __bf16 k0 = (__bf16)kv.x, k1 = (__bf16)kv.y, k2 = (__bf16)kv.z, k3 = (__bf16)kv.w;
        __bf16 v0 = (__bf16)vv.x, v1 = (__bf16)vv.y, v2 = (__bf16)vv.z, v3 = (__bf16)vv.w;
        v4bf kn = {k0, k1, k2, k3};
        v4bf vn = {v0, v1, v2, v3};
        *(v4bf*)&sK[base] = kn;               // natural [c][d]
        *(v4bf*)&sV[base] = vn;
        sKT[(d + 0) * CTILE + c] = k0;        // [d][c]
        sKT[(d + 1) * CTILE + c] = k1;
        sKT[(d + 2) * CTILE + c] = k2;
        sKT[(d + 3) * CTILE + c] = k3;
        sVT[(d + 0) * CTILE + c] = v0;
        sVT[(d + 1) * CTILE + c] = v1;
        sVT[(d + 2) * CTILE + c] = v2;
        sVT[(d + 3) * CTILE + c] = v3;
      }
      if (tid < 32)       sL0[tid]      = glr0[(size_t)bh * SEQ + rowg + tid];
      else if (tid < 64)  sL1[tid - 32] = glr1[(size_t)bh * SEQ + rowg + tid - 32];
      else if (tid < 96)  sL2[tid - 64] = glr2[(size_t)bh * SEQ + rowg + tid - 64];
      if (it + 1 < NITER) {
        __builtin_prefetch((const void*)(kg + 1024), 0, 1);
        __builtin_prefetch((const void*)(vg + 1024), 0, 1);
      }
      __syncthreads();

      // gate / hid_pre / dhidden : 2x8 wave grid over [32 x 128]
      const int mr = (wid >> 3) * 16, nh = (wid & 7) * 16;
      v8f ag, ah, ad;
#pragma unroll
      for (int r = 0; r < 8; ++r) { ag[r] = 0.f; ah[r] = 0.f; ad[r] = 0.f; }
#pragma unroll
      for (int kk = 0; kk < DIM; kk += 32) {
        v16bf a = load_frag(sK, DIM, mr, kk, ln, hi);
        ag = wmma_bf16(a, load_frag(sW0, DIM, nh, kk, ln, hi), ag);
        ah = wmma_bf16(a, load_frag(sW2, DIM, nh, kk, ln, hi), ah);
        ad = wmma_bf16(load_frag(sV, DIM, mr, kk, ln, hi),
                       load_frag(sW1T, DIM, nh, kk, ln, hi), ad);
      }
      // SwiGLU backprop elementwise; lr folded; one b128 store per array
      {
        const int cb = mr + hi * 8;
        const int h = nh + ln;
        float4 l0a = *(const float4*)&sL0[cb], l0b = *(const float4*)&sL0[cb + 4];
        float4 l1a = *(const float4*)&sL1[cb], l1b = *(const float4*)&sL1[cb + 4];
        float4 l2a = *(const float4*)&sL2[cb], l2b = *(const float4*)&sL2[cb + 4];
        float l0v[8] = {l0a.x, l0a.y, l0a.z, l0a.w, l0b.x, l0b.y, l0b.z, l0b.w};
        float l1v[8] = {l1a.x, l1a.y, l1a.z, l1a.w, l1b.x, l1b.y, l1b.z, l1b.w};
        float l2v[8] = {l2a.x, l2a.y, l2a.z, l2a.w, l2b.x, l2b.y, l2b.z, l2b.w};
        v8bf dgv, dpv, htv;
#pragma unroll
        for (int r = 0; r < 8; ++r) {
          float g = ag[r], hp = ah[r], dh = ad[r];
          float sg = 1.f / (1.f + __expf(-g));
          float silu = g * sg;
          float dgate = dh * hp * sg * (1.f + g * (1.f - sg));
          dgv[r] = (__bf16)(dgate * l0v[r]);
          dpv[r] = (__bf16)(dh * silu * l2v[r]);
          htv[r] = (__bf16)(silu * hp * l1v[r]);
        }
        *(v8bf*)&sDgT[h * CTILE + cb] = dgv;
        *(v8bf*)&sDpT[h * CTILE + cb] = dpv;
        *(v8bf*)&sHT [h * CTILE + cb] = htv;
      }
      __syncthreads();

      // dw accumulation: wave owns 4 disjoint 16x16 tiles per matrix.
      // B fragments (kT, vT at this wave's column block) are t-invariant: hoist.
      {
        const int colb = (wid & 7) << 4;
        v16bf bk = load_frag(sKT, CTILE, colb, 0, ln, hi);
        v16bf bv = load_frag(sVT, CTILE, colb, 0, ln, hi);
#pragma unroll
        for (int t = 0; t < 4; ++t) {
          const int orow = (((wid >> 3) << 2) + t) << 4;
          dw0a[t] = wmma_bf16(load_frag(sDgT, CTILE, orow, 0, ln, hi), bk, dw0a[t]);
          dw2a[t] = wmma_bf16(load_frag(sDpT, CTILE, orow, 0, ln, hi), bk, dw2a[t]);
          dw1a[t] = wmma_bf16(load_frag(sHT,  CTILE, orow, 0, ln, hi), bv, dw1a[t]);
        }
      }
      __syncthreads();
    }

    // ======== chunk end: stage dw, momentum + L2 renorm ========
#pragma unroll
    for (int t = 0; t < 4; ++t) {
      const int orow = (((wid >> 3) << 2) + t) << 4;
      const int colb = (wid & 7) << 4;
#pragma unroll
      for (int r = 0; r < 8; ++r) {
        int row = orow + hi * 8 + r, cc = colb + ln;
        st0[row * DIM + cc] = dw0a[t][r];
        st1[row * DIM + cc] = dw1a[t][r];
        st2[row * DIM + cc] = dw2a[t][r];
      }
    }
    __threadfence();
    __syncthreads();
    {
      const float mm = sMM[0];
      const int row = tid >> 2, cb = (tid & 3) * 32;
      update_w(w0m, m0p, st0, false, sN0, sW0,  false, mm, ch == 0, row, cb, sRed, tid);
      __syncthreads();
      update_w(w1m, m1p, st1, true,  sN1, sW1T, true,  mm, ch == 0, row, cb, sRed, tid);
      __syncthreads();
      update_w(w2m, m2p, st2, false, sN2, sW2,  false, mm, ch == 0, row, cb, sRed, tid);
      __syncthreads();
    }

    // ======== Q pass ========
    // Hoist per-wave w1 A-fragments (o^T matmul) from updated fp32 master.
    const int dr = (wid >> 1) * 16, c0q = (wid & 1) * 16;
    v16bf w1A[4];
    {
      const float* wrow = w1m + (size_t)(dr + ln) * DIM + hi * 8;
#pragma unroll
      for (int kq = 0; kq < 4; ++kq) {
        const float* rr = wrow + kq * 32;
#pragma unroll
        for (int e = 0; e < 8; ++e) {
          w1A[kq][e]     = (__bf16)rr[e];
          w1A[kq][8 + e] = (__bf16)rr[16 + e];
        }
      }
    }

    for (int it = 0; it < NITER; ++it) {
      const int rowg = ch * CHUNK + it * CTILE;
      const float4* qg = (const float4*)(gq + ((size_t)bh * SEQ + rowg) * DIM);
#pragma unroll
      for (int j = 0; j < 2; ++j) {
        int f4 = tid + j * NTHR;
        float4 qv = qg[f4];
        int base = f4 * 4;
        v4bf qn = {(__bf16)qv.x, (__bf16)qv.y, (__bf16)qv.z, (__bf16)qv.w};
        *(v4bf*)&sK[base] = qn;
      }
      if (it + 1 < NITER) __builtin_prefetch((const void*)(qg + 1024), 0, 1);
      __syncthreads();

      const int mr = (wid >> 3) * 16, nh = (wid & 7) * 16;
      v8f ag, ah;
#pragma unroll
      for (int r = 0; r < 8; ++r) { ag[r] = 0.f; ah[r] = 0.f; }
#pragma unroll
      for (int kk = 0; kk < DIM; kk += 32) {
        v16bf a = load_frag(sK, DIM, mr, kk, ln, hi);
        ag = wmma_bf16(a, load_frag(sW0, DIM, nh, kk, ln, hi), ag);
        ah = wmma_bf16(a, load_frag(sW2, DIM, nh, kk, ln, hi), ah);
      }
      // h natural [c][h] (B N-major for o^T)
#pragma unroll
      for (int r = 0; r < 8; ++r) {
        int c = mr + hi * 8 + r;
        int h = nh + ln;
        float g = ag[r];
        float sg = 1.f / (1.f + __expf(-g));
        sV[c * DIM + h] = (__bf16)(g * sg * ah[r]);
      }
      __syncthreads();

      // o^T[d][c] = sum_h w1[d][h] * h[c][h]^T  (A = w1 regs, B = sV N-major)
      v8f ao;
#pragma unroll
      for (int r = 0; r < 8; ++r) ao[r] = 0.f;
#pragma unroll
      for (int kk = 0; kk < DIM; kk += 32)
        ao = wmma_bf16(w1A[kk >> 5], load_frag(sV, DIM, c0q, kk, ln, hi), ao);
#pragma unroll
      for (int r = 0; r < 8; ++r)
        sOT[(dr + hi * 8 + r) * CTILE + c0q + ln] = ao[r];
      __syncthreads();

      // coalesced fp32 store of o [32 x 128]
      float* og = gout + ((size_t)bh * SEQ + rowg) * DIM;
#pragma unroll
      for (int j = 0; j < 2; ++j) {
        int f4 = tid + j * NTHR;
        int base = f4 * 4;
        int c = base >> 7, d = base & 127;
        float4 o4;
        o4.x = sOT[(d + 0) * CTILE + c];
        o4.y = sOT[(d + 1) * CTILE + c];
        o4.z = sOT[(d + 2) * CTILE + c];
        o4.w = sOT[(d + 3) * CTILE + c];
        ((float4*)og)[f4] = o4;
      }
      __syncthreads();
    }
  }
}

extern "C" void kernel_launch(void* const* d_in, const int* in_sizes, int n_in,
                              void* d_out, int out_size, void* d_ws, size_t ws_size,
                              hipStream_t stream) {
  const float* w0  = (const float*)d_in[0];
  const float* w1  = (const float*)d_in[1];
  const float* w2  = (const float*)d_in[2];
  const float* q   = (const float*)d_in[3];
  const float* k   = (const float*)d_in[4];
  const float* v   = (const float*)d_in[5];
  const float* lr0 = (const float*)d_in[6];
  const float* lr1 = (const float*)d_in[7];
  const float* lr2 = (const float*)d_in[8];
  const float* mom = (const float*)d_in[9];
  float* out = (float*)d_out;
  float* ws  = (float*)d_ws;
  lact_swiglu_kernel<<<dim3(BH_N), dim3(NTHR), 0, stream>>>(
      w0, w1, w2, q, k, v, lr0, lr1, lr2, mom, out, ws);
}
